// one_batch_net_9904194585140
// MI455X (gfx1250) — matrix operand order repaired and checked by reading, not compiled, not run on hardware
//
#include <hip/hip_runtime.h>
#include <hip/hip_bf16.h>

typedef __attribute__((ext_vector_type(16))) _Float16 v16h;
typedef __attribute__((ext_vector_type(8)))  _Float16 v8h;
typedef __attribute__((ext_vector_type(8)))  float    v8f;

namespace {

constexpr int TT   = 4;     // atom types
constexpr int NB   = 8;     // batches
constexpr int NA   = 512;   // atoms per batch
constexpr int NS   = 128;   // neighbors
constexpr int F0   = 32;
constexpr int F1   = 64;
constexpr int F2   = 128;
constexpr int HH   = 256;
constexpr int DESC = 2048;  // 16 * 128
constexpr int NATOMS = NB * NA;  // 4096

// tanh via v_exp_f32 + v_rcp_f32 (avoid IEEE-precise division expansion):
// tanh(x) = (e^{2x} - 1) * rcp(e^{2x} + 1); clamp folds into v_med3_num_f32.
__device__ __forceinline__ float fast_tanh(float x) {
  x = fminf(15.f, fmaxf(-15.f, x));
  float e = __expf(2.f * x);
  return (e - 1.f) * __builtin_amdgcn_rcpf(e + 1.f);
}

__device__ __forceinline__ v16h cat16(v8h lo, v8h hi) {
  return __builtin_shufflevector(lo, hi, 0,1,2,3,4,5,6,7,8,9,10,11,12,13,14,15);
}

// WMMA A operand: 16 rows x 32 K fp16 slab at column k0 of a row-major fp16
// matrix with leading dim `ld`.  Per ISA: lanes 0-15 row=lane hold K k0..k0+7
// and k0+16..k0+23; lanes 16-31 row=lane-16 hold k0+8..k0+15 and k0+24..k0+31.
__device__ __forceinline__ v16h load_a_tile(const _Float16* base, int ld, int lane, int k0) {
  const _Float16* p = base + (lane & 15) * ld + k0 + ((lane >> 4) << 3);
  return cat16(*(const v8h*)p, *(const v8h*)(p + 16));
}

// WMMA B operand: B[k][n] = W[n0+n][k0+k], W row-major (out,in) with leading
// dim `ld`; B column n == contiguous chunk of W row n0+n.  Lanes 0-15 hold
// K k0..k0+7 and k0+16..k0+23 of column lane, lanes 16-31 hold K k0+8..k0+15
// and k0+24..k0+31.
__device__ __forceinline__ v16h load_b_tile(const _Float16* wgt, int ld, int n0, int k0, int lane) {
  const _Float16* p = wgt + (size_t)(n0 + (lane & 15)) * ld + k0 + ((lane >> 4) << 3);
  return cat16(*(const v8h*)p, *(const v8h*)(p + 16));
}

__device__ __forceinline__ v8f wmma_f16(v16h a, v16h b, v8f c) {
  return __builtin_amdgcn_wmma_f32_16x16x32_f16(false, a, false, b, (short)0, c, false, false);
}

// ---------------------------------------------------------------- prep ----
__global__ void cvt_f16_kernel(const float* __restrict__ src, _Float16* __restrict__ dst, int n) {
  int i = blockIdx.x * blockDim.x + threadIdx.x;
  if (i < n) dst[i] = (_Float16)src[i];
}

__global__ void bucket_kernel(const int* __restrict__ types, int* counts, int* bucket) {
  int i = blockIdx.x * blockDim.x + threadIdx.x;
  if (i < NATOMS) {
    int t = types[i];
    int pos = atomicAdd(&counts[t], 1);
    bucket[t * NATOMS + pos] = i;
  }
}

// -------------------------------------------------- filter net + descriptor
// One workgroup (4 waves) per atom.
__global__ void __launch_bounds__(128) filter_kernel(
    const float* __restrict__ sym, const int* __restrict__ types,
    const float* __restrict__ Wf0, const float* __restrict__ bf0,
    const float* __restrict__ bf1, const float* __restrict__ bf2,
    const _Float16* __restrict__ Wf1h, const _Float16* __restrict__ Wf2h,
    _Float16* __restrict__ Dh)
{
  // Hand-packed LDS arena (strides padded +8 halves -> conflict-free b128):
  //   [0,10240)      sG0  128 x 40 fp16      (dead after GEMM1; reused below)
  //   [10240,28672)  sG1  128 x 72 fp16
  //   [28672,63488)  sG2T 128 x 136 fp16  (TRANSPOSED: [m][s])
  //   reuse of sG0 region after GEMM2:
  //   [0,4352)       sCA  16 x 136 fp16   (coord A-matrix, rows 4..15 zero)
  //   [4352,6400)    sRG  4 x 128 f32
  __shared__ __align__(16) char smem[63488];
  _Float16* sG0  = (_Float16*)smem;
  _Float16* sG1  = (_Float16*)(smem + 10240);
  _Float16* sG2T = (_Float16*)(smem + 28672);
  _Float16* sCA  = (_Float16*)smem;
  float*    sRG  = (float*)(smem + 4352);

  const int atom = blockIdx.x;
  const int tid  = threadIdx.x;
  const int lane = tid & 31;
  const int w    = tid >> 5;
  const int t    = types[atom];

  // layer f0: G0[s][f] = tanh(s * Wf0[t][f] + bf0[t][f])   (K == 1 -> VALU)
  {
    float sv = sym[(size_t)atom * (NS * 4) + tid * 4];
    const float* w0 = Wf0 + t * F0;
    const float* b0 = bf0 + t * F0;
#pragma unroll 4
    for (int f = 0; f < F0; ++f)
      sG0[tid * 40 + f] = (_Float16)fast_tanh(sv * w0[f] + b0[f]);
  }
  __syncthreads();

  // GEMM1: G1 = tanh(G0(128x32) @ Wf1[t]^T(32x64) + bf1)   8x4 tiles
  {
    const _Float16* wB  = Wf1h + t * (F1 * F0);
    const float* bias   = bf1 + t * F1;
    for (int id = w; id < 32; id += 4) {
      int mt = id >> 2, nt = id & 3;
      v16h a = load_a_tile(sG0 + mt * 16 * 40, 40, lane, 0);
      v16h b = load_b_tile(wB, F0, nt * 16, 0, lane);
      v8f  c = {};
      c = wmma_f16(a, b, c);
      int n = lane & 15, rb = (lane >> 4) << 3;
      float bv = bias[nt * 16 + n];
#pragma unroll
      for (int j = 0; j < 8; ++j)
        sG1[(mt * 16 + rb + j) * 72 + nt * 16 + n] = (_Float16)fast_tanh(c[j] + bv);
    }
  }
  __syncthreads();

  // GEMM2: G2 = tanh(G1(128x64) @ Wf2[t]^T(64x128) + bf2)  8x8 tiles, K=64
  // stored TRANSPOSED so the RG GEMM's B columns are contiguous in s.
  {
    const _Float16* wB  = Wf2h + t * (F2 * F1);
    const float* bias   = bf2 + t * F2;
    for (int id = w; id < 64; id += 4) {
      int mt = id >> 3, nt = id & 7;
      v8f c = {};
#pragma unroll
      for (int k0 = 0; k0 < F1; k0 += 32) {
        v16h a = load_a_tile(sG1 + mt * 16 * 72, 72, lane, k0);
        v16h b = load_b_tile(wB, F1, nt * 16, k0, lane);
        c = wmma_f16(a, b, c);
      }
      int n = lane & 15, rb = (lane >> 4) << 3;
      float bv = bias[nt * 16 + n];
#pragma unroll
      for (int j = 0; j < 8; ++j)
        sG2T[(nt * 16 + n) * 136 + mt * 16 + rb + j] = (_Float16)fast_tanh(c[j] + bv);
    }
  }
  __syncthreads();

  // build coord A-matrix: rows 0..3 = coord dims d over s, rows 4..15 zero
  {
    float4 c4 = *(const float4*)(sym + (size_t)atom * (NS * 4) + tid * 4);
    sCA[0 * 136 + tid] = (_Float16)c4.x;
    sCA[1 * 136 + tid] = (_Float16)c4.y;
    sCA[2 * 136 + tid] = (_Float16)c4.z;
    sCA[3 * 136 + tid] = (_Float16)c4.w;
#pragma unroll
    for (int r = 4; r < 16; ++r) sCA[r * 136 + tid] = (_Float16)0.f;
  }
  __syncthreads();

  // RG(4x128) = coordA(16x128) @ G2(128x128); B tiles read from sG2T rows.
  for (int nt = w; nt < 8; nt += 4) {
    v8f c = {};
#pragma unroll
    for (int k0 = 0; k0 < 128; k0 += 32) {
      v16h a = load_a_tile(sCA, 136, lane, k0);
      v16h b = load_b_tile(sG2T, 136, nt * 16, k0, lane);
      c = wmma_f16(a, b, c);
    }
    int n = lane & 15, rb = (lane >> 4) << 3;
#pragma unroll
    for (int j = 0; j < 8; ++j) {
      int d = rb + j;
      if (d < 4) sRG[d * 128 + nt * 16 + n] = c[j];
    }
  }
  __syncthreads();

  // D[m*16+a] = sum_d RG[d][m] * RG[d][a]  -> fp16 to workspace (coalesced)
  {
    _Float16* dOut = Dh + (size_t)atom * DESC;
#pragma unroll 4
    for (int r = 0; r < 16; ++r) {
      int idx = r * 128 + tid;
      int m = idx >> 4, a = idx & 15;
      float v = sRG[0 * 128 + m] * sRG[0 * 128 + a]
              + sRG[1 * 128 + m] * sRG[1 * 128 + a]
              + sRG[2 * 128 + m] * sRG[2 * 128 + a]
              + sRG[3 * 128 + m] * sRG[3 * 128 + a];
      dOut[idx] = (_Float16)v;
    }
  }
}

// ------------------------------------------------------------- fit net ----
// One layer: h_out(16xHH) = tanh(A(16xK) @ W[t]^T(KxHH) + bias), fp16 out.
// Each wave owns 4 N-tiles (n0 = (w*4+q)*16); A row ptr is per-lane.
__device__ __forceinline__ void fit_layer_h16(
    const _Float16* aRow, const _Float16* wB, int K,
    const float* bias, _Float16* dstH, int lane, int w)
{
  v8f acc[4] = {};
  for (int k0 = 0; k0 < K; k0 += 32) {
    v16h a = cat16(*(const v8h*)(aRow + k0), *(const v8h*)(aRow + k0 + 16));
#pragma unroll
    for (int q = 0; q < 4; ++q)
      acc[q] = wmma_f16(a, load_b_tile(wB, K, (w * 4 + q) * 16, k0, lane), acc[q]);
  }
  int n = lane & 15, rb = (lane >> 4) << 3;
#pragma unroll
  for (int q = 0; q < 4; ++q) {
    int n0 = (w * 4 + q) * 16;
    float bv = bias[n0 + n];
#pragma unroll
    for (int j = 0; j < 8; ++j)
      dstH[(rb + j) * 264 + n0 + n] = (_Float16)fast_tanh(acc[q][j] + bv);
  }
}

// 16 same-type atoms per workgroup; grid = TT * (NATOMS/16).
__global__ void __launch_bounds__(128) fit_kernel(
    const _Float16* __restrict__ Dh, const int* __restrict__ bucket,
    const int* __restrict__ counts,
    const _Float16* __restrict__ Wt0h, const _Float16* __restrict__ Wt1h,
    const _Float16* __restrict__ Wt2h,
    const float* __restrict__ bt0, const float* __restrict__ bt1,
    const float* __restrict__ bt2,
    const float* __restrict__ Wo, const float* __restrict__ bo,
    float* __restrict__ Eatom)
{
  __shared__ int      sAtoms[16];
  __shared__ _Float16 sHa[16 * 264];
  __shared__ _Float16 sHb[16 * 264];
  __shared__ float    sH3[16 * 264];
  __shared__ float    sRed[16 * 8];

  const int t    = blockIdx.x >> 8;       // NATOMS/16 == 256 groups per type
  const int g    = blockIdx.x & 255;
  const int cnt  = counts[t];
  const int base = g * 16;
  if (base >= cnt) return;                // block-uniform: whole WG exits
  const int nval = min(16, cnt - base);

  const int tid = threadIdx.x, lane = tid & 31, w = tid >> 5;
  if (tid < 16) sAtoms[tid] = bucket[t * NATOMS + base + min(tid, nval - 1)];
  __syncthreads();

  const int aoff = (lane >> 4) << 3;      // per-lane K sub-offset of A layout

  // layer t0: K = 2048, A streamed from global descriptor rows (L2-hot)
  {
    const _Float16* aRow = Dh + (size_t)sAtoms[lane & 15] * DESC + aoff;
    fit_layer_h16(aRow, Wt0h + (size_t)t * HH * DESC, DESC, bt0 + t * HH, sHa, lane, w);
  }
  __syncthreads();
  // layer t1: K = 256, A from LDS
  {
    const _Float16* aRow = sHa + (lane & 15) * 264 + aoff;
    fit_layer_h16(aRow, Wt1h + (size_t)t * HH * HH, HH, bt1 + t * HH, sHb, lane, w);
  }
  __syncthreads();
  // layer t2: K = 256, f32 output for the final dot product
  {
    const _Float16* aRow = sHb + (lane & 15) * 264 + aoff;
    const _Float16* wB   = Wt2h + (size_t)t * HH * HH;
    const float* bias    = bt2 + t * HH;
    v8f acc[4] = {};
    for (int k0 = 0; k0 < HH; k0 += 32) {
      v16h a = cat16(*(const v8h*)(aRow + k0), *(const v8h*)(aRow + k0 + 16));
#pragma unroll
      for (int q = 0; q < 4; ++q)
        acc[q] = wmma_f16(a, load_b_tile(wB, HH, (w * 4 + q) * 16, k0, lane), acc[q]);
    }
    int n = lane & 15, rb = (lane >> 4) << 3;
#pragma unroll
    for (int q = 0; q < 4; ++q) {
      int n0 = (w * 4 + q) * 16;
      float bv = bias[n0 + n];
#pragma unroll
      for (int j = 0; j < 8; ++j)
        sH3[(rb + j) * 264 + n0 + n] = fast_tanh(acc[q][j] + bv);
    }
  }
  __syncthreads();

  // E[m] = sum_h h3[m][h] * Wo[t][h] + bo[t]   (8 partials per atom)
  {
    int m = tid >> 3, p = tid & 7;
    const float* wv = Wo + t * HH;
    float acc = 0.f;
#pragma unroll 8
    for (int h = p * 32; h < p * 32 + 32; ++h) acc += sH3[m * 264 + h] * wv[h];
    sRed[m * 8 + p] = acc;
  }
  __syncthreads();
  if (tid < 16) {
    float e = bo[t];
#pragma unroll
    for (int p = 0; p < 8; ++p) e += sRed[tid * 8 + p];
    if (tid < nval) Eatom[sAtoms[tid]] = e;    // padded rows masked out
  }
}

// ------------------------------------------------- deterministic reduce ---
__global__ void reduce_kernel(const float* __restrict__ Eatom, float* __restrict__ out) {
  __shared__ float sbuf[256];
  int b = blockIdx.x, tid = threadIdx.x;
  sbuf[tid] = Eatom[b * NA + tid] + Eatom[b * NA + 256 + tid];
  __syncthreads();
  for (int s = 128; s > 0; s >>= 1) {
    if (tid < s) sbuf[tid] += sbuf[tid + s];
    __syncthreads();
  }
  if (tid == 0) out[b] = sbuf[0];
}

}  // namespace

extern "C" void kernel_launch(void* const* d_in, const int* in_sizes, int n_in,
                              void* d_out, int out_size, void* d_ws, size_t ws_size,
                              hipStream_t stream) {
  const float* sym   = (const float*)d_in[0];
  const int*   types = (const int*)d_in[1];
  const float* Wf0 = (const float*)d_in[2];
  const float* bf0 = (const float*)d_in[3];
  const float* Wf1 = (const float*)d_in[4];
  const float* bf1 = (const float*)d_in[5];
  const float* Wf2 = (const float*)d_in[6];
  const float* bf2 = (const float*)d_in[7];
  const float* Wt0 = (const float*)d_in[8];
  const float* bt0 = (const float*)d_in[9];
  const float* Wt1 = (const float*)d_in[10];
  const float* bt1 = (const float*)d_in[11];
  const float* Wt2 = (const float*)d_in[12];
  const float* bt2 = (const float*)d_in[13];
  const float* Wo  = (const float*)d_in[14];
  const float* bo  = (const float*)d_in[15];
  float* out = (float*)d_out;

  char* ws = (char*)d_ws;
  size_t o = 0;
  auto take = [&](size_t bytes) -> char* {
    char* p = ws + o;
    o = (o + bytes + 255) & ~(size_t)255;
    return p;
  };
  _Float16* Wf1h = (_Float16*)take((size_t)TT * F1 * F0 * 2);
  _Float16* Wf2h = (_Float16*)take((size_t)TT * F2 * F1 * 2);
  _Float16* Wt0h = (_Float16*)take((size_t)TT * HH * DESC * 2);
  _Float16* Wt1h = (_Float16*)take((size_t)TT * HH * HH * 2);
  _Float16* Wt2h = (_Float16*)take((size_t)TT * HH * HH * 2);
  _Float16* Dh   = (_Float16*)take((size_t)NATOMS * DESC * 2);
  int*   counts  = (int*)take(TT * sizeof(int));
  int*   bucket  = (int*)take((size_t)TT * NATOMS * sizeof(int));
  float* Eatom   = (float*)take((size_t)NATOMS * sizeof(float));

  hipMemsetAsync(counts, 0, TT * sizeof(int), stream);

  cvt_f16_kernel<<<(TT * F1 * F0 + 255) / 256, 256, 0, stream>>>(Wf1, Wf1h, TT * F1 * F0);
  cvt_f16_kernel<<<(TT * F2 * F1 + 255) / 256, 256, 0, stream>>>(Wf2, Wf2h, TT * F2 * F1);
  cvt_f16_kernel<<<(TT * HH * DESC + 255) / 256, 256, 0, stream>>>(Wt0, Wt0h, TT * HH * DESC);
  cvt_f16_kernel<<<(TT * HH * HH + 255) / 256, 256, 0, stream>>>(Wt1, Wt1h, TT * HH * HH);
  cvt_f16_kernel<<<(TT * HH * HH + 255) / 256, 256, 0, stream>>>(Wt2, Wt2h, TT * HH * HH);

  bucket_kernel<<<NATOMS / 256, 256, 0, stream>>>(types, counts, bucket);

  filter_kernel<<<NATOMS, 128, 0, stream>>>(sym, types, Wf0, bf0, bf1, bf2, Wf1h, Wf2h, Dh);

  fit_kernel<<<TT * (NATOMS / 16), 128, 0, stream>>>(Dh, bucket, counts,
      Wt0h, Wt1h, Wt2h, bt0, bt1, bt2, Wo, bo, Eatom);

  reduce_kernel<<<NB, 256, 0, stream>>>(Eatom, out);
}
